// InteractionNetworkLayer_23450521436276
// MI455X (gfx1250) — matrix operand order repaired and checked by reading, not compiled, not run on hardware
//
#include <hip/hip_runtime.h>
#include <hip/hip_bf16.h>
#include <math.h>

#define NN 50000
#define NE 500000
#define DD 128

typedef __bf16 v16bf __attribute__((ext_vector_type(16)));
typedef __bf16 v2bf  __attribute__((ext_vector_type(2)));
typedef float  v8f   __attribute__((ext_vector_type(8)));
typedef float  v2f   __attribute__((ext_vector_type(2)));
typedef unsigned int v4u __attribute__((ext_vector_type(4)));

union Frag { v16bf v; unsigned int u[8]; v4u q[2]; };

__device__ __forceinline__ unsigned short f2bf(float f) {
  unsigned int u = __float_as_uint(f);
  u += 0x7FFFu + ((u >> 16) & 1u);   // round-to-nearest-even (prep path only)
  return (unsigned short)(u >> 16);
}

#if defined(__has_builtin)
#  if __has_builtin(__builtin_amdgcn_cvt_pk_bf16_f32)
#    define USE_CVT_PK 1
#  endif
#endif

#ifdef USE_CVT_PK
__device__ __forceinline__ unsigned int pack2bf(float a, float b) {
  union { v2bf v; unsigned int u; } c;
  c.v = __builtin_amdgcn_cvt_pk_bf16_f32(a, b);
  return c.u;
}
#else
// round-half-up then single v_perm_b32 packs both high halves
__device__ __forceinline__ unsigned int pack2bf(float a, float b) {
  unsigned int ua = __float_as_uint(a) + 0x8000u;
  unsigned int ub = __float_as_uint(b) + 0x8000u;
  return __builtin_amdgcn_perm(ub, ua, 0x07060302u);
}
#endif

__device__ __forceinline__ unsigned short f2bf_fast(float f) {
  return (unsigned short)((__float_as_uint(f) + 0x8000u) >> 16);  // round-half-up
}

// Weight scratch layout inside d_ws (after the adjacency accumulator):
//   eW1T [128n][384k] | eW2T [128n][128k] | nW1T [128n][256k] | nW2T [128n][128k]
#define ADJ_FLOATS  ((size_t)NN * DD)
#define EW1T_ELEMS  (384 * 128)
#define EW2T_ELEMS  (128 * 128)
#define NW1T_ELEMS  (256 * 128)
#define NW2T_ELEMS  (128 * 128)

// ---------------------------------------------------------------------------
// One-time: convert weights to bf16, transposed to n-major, into d_ws.
// ---------------------------------------------------------------------------
__global__ __launch_bounds__(256)
void prep_weights_kernel(const float* __restrict__ eW1, const float* __restrict__ eW2,
                         const float* __restrict__ nW1, const float* __restrict__ nW2,
                         unsigned short* __restrict__ wbf)
{
  unsigned short* eW1T = wbf;
  unsigned short* eW2T = eW1T + EW1T_ELEMS;
  unsigned short* nW1T = eW2T + EW2T_ELEMS;
  unsigned short* nW2T = nW1T + NW1T_ELEMS;
  const int t = blockIdx.x * blockDim.x + threadIdx.x;
  const int stride = gridDim.x * blockDim.x;
  for (int i = t; i < EW1T_ELEMS; i += stride) { int k = i >> 7, n = i & 127; eW1T[n * 384 + k] = f2bf(eW1[i]); }
  for (int i = t; i < EW2T_ELEMS; i += stride) { int k = i >> 7, n = i & 127; eW2T[n * 128 + k] = f2bf(eW2[i]); }
  for (int i = t; i < NW1T_ELEMS; i += stride) { int k = i >> 7, n = i & 127; nW1T[n * 256 + k] = f2bf(nW1[i]); }
  for (int i = t; i < NW2T_ELEMS; i += stride) { int k = i >> 7, n = i & 127; nW2T[n * 128 + k] = f2bf(nW2[i]); }
}

// Build one 16x32 bf16 A-fragment from an f32 row (ISA 16-bit A layout).
__device__ __forceinline__ Frag load_a_frag(const float* abase, int half) {
  Frag a;
#pragma unroll
  for (int v = 0; v < 8; ++v) {
    int ko = ((v & 4) << 2) + half * 8 + ((v & 3) << 1);  // (v>=4?16:0)+half*8+(v&3)*2
    v2f f = *(const v2f*)(abase + ko);
    a.u[v] = pack2bf(f.x, f.y);
  }
  return a;
}

// ---------------------------------------------------------------------------
// Edge MLP:  e_upd = relu([edges|sent|recv] @ eW1 + eb1) @ eW2 + eb2
//            edges_out = edges + mult*e_upd ;  adj[recv] += e_upd (atomic)
// 256 threads = 8 waves = 4 M-groups x 2 N-groups; wave tile = 32 rows x 64 cols.
// Block tile = 128 edges x 128 outputs.
// ---------------------------------------------------------------------------
__global__ __launch_bounds__(256)
void edge_mlp_kernel(const float* __restrict__ nodes,
                     const float* __restrict__ edges,
                     const int*   __restrict__ receivers,
                     const int*   __restrict__ senders,
                     const unsigned short* __restrict__ wbf,
                     const float* __restrict__ eb1,
                     const float* __restrict__ eb2,
                     const float* __restrict__ rm_w,
                     float* __restrict__ edges_out,
                     float* __restrict__ adj)
{
  extern __shared__ __align__(16) unsigned short smem[];
  unsigned short* sW1 = smem;                    // [128 n][384 k]
  unsigned short* sW2 = smem + 128 * 384;        // [128 n][128 k]
  unsigned short* sH  = sW2  + 128 * 128;        // [128 m][128 k]

  const int tid = threadIdx.x;

  // stage pre-converted bf16 weights: wide copies only
  {
    const v4u* g1 = (const v4u*)wbf;
    v4u* s1 = (v4u*)sW1;
    for (int i = tid; i < EW1T_ELEMS / 8; i += 256) s1[i] = g1[i];
    const v4u* g2 = (const v4u*)(wbf + EW1T_ELEMS);
    v4u* s2 = (v4u*)sW2;
    for (int i = tid; i < EW2T_ELEMS / 8; i += 256) s2[i] = g2[i];
  }
  __syncthreads();

  const int wid  = tid >> 5;
  const int lane = tid & 31;
  const int l16  = lane & 15;
  const int half = lane >> 4;
  const int mg   = wid >> 1;                 // 0..3  (M group, 32 rows)
  const int cg   = wid & 1;                  // 0..1  (N group, 64 cols)
  const int mbase = blockIdx.x * 128 + mg * 32;

  // two 16-row A fragments per wave
  const float* ap[2][3];
#pragma unroll
  for (int mf = 0; mf < 2; ++mf) {
    int erow = mbase + mf * 16 + l16;
    int ec   = erow < NE ? erow : NE - 1;
    ap[mf][0] = edges + (size_t)ec * DD;                 // k in [0,128)
    ap[mf][1] = nodes + (size_t)senders[ec]   * DD;      // k in [128,256)
    ap[mf][2] = nodes + (size_t)receivers[ec] * DD;      // k in [256,384)
  }

  // -------- GEMM1: [128 x 384] x [384 x 128], software pipelined --------
  v8f acc[2][4] = {};
  Frag a_cur[2] = { load_a_frag(ap[0][0], half), load_a_frag(ap[1][0], half) };
  for (int kp = 0; kp < 12; ++kp) {
    // 4 B fragments (8x ds_load_b128), shared by both M fragments
    Frag b[4];
    const unsigned short* wrow = sW1 + (cg * 64 + l16) * 384 + kp * 32 + half * 8;
#pragma unroll
    for (int j = 0; j < 4; ++j) {
      const v4u* wp = (const v4u*)(wrow + j * (16 * 384));
      b[j].q[0] = wp[0];
      b[j].q[1] = wp[2];
    }
    // prefetch next K-step's A fragments (global) to hide latency
    Frag a_next[2];
    if (kp < 11) {
      int kn = kp + 1;
      int r  = kn >> 2;
#pragma unroll
      for (int mf = 0; mf < 2; ++mf) {
        const float* abase = (r == 0) ? ap[mf][0] : (r == 1 ? ap[mf][1] : ap[mf][2]);
        a_next[mf] = load_a_frag(abase + (kn & 3) * 32, half);
      }
    }
#pragma unroll
    for (int mf = 0; mf < 2; ++mf)
#pragma unroll
      for (int j = 0; j < 4; ++j)
        acc[mf][j] = __builtin_amdgcn_wmma_f32_16x16x32_bf16(
            false, a_cur[mf].v, false, b[j].v, (short)0, acc[mf][j], false, false);
    a_cur[0] = a_next[0];
    a_cur[1] = a_next[1];
  }

  // bias + relu -> bf16 hidden tile
#pragma unroll
  for (int mf = 0; mf < 2; ++mf)
#pragma unroll
    for (int j = 0; j < 4; ++j) {
      int col = cg * 64 + j * 16 + l16;
      float b1v = eb1[col];
#pragma unroll
      for (int v = 0; v < 8; ++v) {
        float x = acc[mf][j][v] + b1v;
        x = x > 0.f ? x : 0.f;
        int m = mg * 32 + mf * 16 + half * 8 + v;
        sH[m * 128 + col] = f2bf_fast(x);
      }
    }
  __syncthreads();   // H columns now span both N-groups

  // -------- GEMM2: [128 x 128] x [128 x 128] --------
  v8f acc2[2][4] = {};
  const unsigned short* hrow0 = sH + (mg * 32 + l16) * 128;
  const unsigned short* hrow1 = hrow0 + 16 * 128;
  for (int kp = 0; kp < 4; ++kp) {
    Frag b[4];
    const unsigned short* wrow = sW2 + (cg * 64 + l16) * 128 + kp * 32 + half * 8;
#pragma unroll
    for (int j = 0; j < 4; ++j) {
      const v4u* wp = (const v4u*)(wrow + j * (16 * 128));
      b[j].q[0] = wp[0];
      b[j].q[1] = wp[2];
    }
    Frag a[2];
#pragma unroll
    for (int v = 0; v < 8; ++v) {
      int ko = kp * 32 + ((v & 4) << 2) + half * 8 + ((v & 3) << 1);
      a[0].u[v] = *(const unsigned int*)(hrow0 + ko);
      a[1].u[v] = *(const unsigned int*)(hrow1 + ko);
    }
#pragma unroll
    for (int mf = 0; mf < 2; ++mf)
#pragma unroll
      for (int j = 0; j < 4; ++j)
        acc2[mf][j] = __builtin_amdgcn_wmma_f32_16x16x32_bf16(
            false, a[mf].v, false, b[j].v, (short)0, acc2[mf][j], false, false);
  }

  // -------- epilogue: residual write + atomic scatter-sum --------
  float mult = log1pf(expf(rm_w[0]));   // softplus
  int rcv[2][8];
#pragma unroll
  for (int mf = 0; mf < 2; ++mf)
#pragma unroll
    for (int v = 0; v < 8; ++v) {
      int e = mbase + mf * 16 + half * 8 + v;
      rcv[mf][v] = (e < NE) ? receivers[e] : -1;
    }
#pragma unroll
  for (int mf = 0; mf < 2; ++mf)
#pragma unroll
    for (int j = 0; j < 4; ++j) {
      int col = cg * 64 + j * 16 + l16;
      float b2v = eb2[col];
#pragma unroll
      for (int v = 0; v < 8; ++v) {
        if (rcv[mf][v] >= 0) {
          int e = mbase + mf * 16 + half * 8 + v;
          float val = acc2[mf][j][v] + b2v;            // e_upd
          size_t off = (size_t)e * DD + col;
          edges_out[off] = edges[off] + mult * val;
          atomicAdd(adj + (size_t)rcv[mf][v] * DD + col, val);
        }
      }
    }
}

// ---------------------------------------------------------------------------
// Node MLP:  n_upd = relu([nodes|adj] @ nW1 + nb1) @ nW2 + nb2
//            nodes_out = nodes + mult*n_upd
// ---------------------------------------------------------------------------
__global__ __launch_bounds__(256)
void node_mlp_kernel(const float* __restrict__ nodes,
                     const float* __restrict__ adj,
                     const unsigned short* __restrict__ wbf,
                     const float* __restrict__ nb1,
                     const float* __restrict__ nb2,
                     const float* __restrict__ rm_w,
                     float* __restrict__ nodes_out)
{
  extern __shared__ __align__(16) unsigned short smem[];
  unsigned short* sW1 = smem;                    // [128 n][256 k]
  unsigned short* sW2 = smem + 128 * 256;        // [128 n][128 k]
  unsigned short* sH  = sW2  + 128 * 128;        // [128 m][128 k]

  const int tid = threadIdx.x;
  {
    const unsigned short* gw1 = wbf + EW1T_ELEMS + EW2T_ELEMS;
    const v4u* g1 = (const v4u*)gw1;
    v4u* s1 = (v4u*)sW1;
    for (int i = tid; i < NW1T_ELEMS / 8; i += 256) s1[i] = g1[i];
    const v4u* g2 = (const v4u*)(gw1 + NW1T_ELEMS);
    v4u* s2 = (v4u*)sW2;
    for (int i = tid; i < NW2T_ELEMS / 8; i += 256) s2[i] = g2[i];
  }
  __syncthreads();

  const int wid  = tid >> 5;
  const int lane = tid & 31;
  const int l16  = lane & 15;
  const int half = lane >> 4;
  const int mg   = wid >> 1;
  const int cg   = wid & 1;
  const int mbase = blockIdx.x * 128 + mg * 32;

  const float* ap[2][2];
#pragma unroll
  for (int mf = 0; mf < 2; ++mf) {
    int nrow = mbase + mf * 16 + l16;
    int nc   = nrow < NN ? nrow : NN - 1;
    ap[mf][0] = nodes + (size_t)nc * DD;   // k in [0,128)
    ap[mf][1] = adj   + (size_t)nc * DD;   // k in [128,256)
  }

  // -------- GEMM1: [128 x 256] x [256 x 128], software pipelined --------
  v8f acc[2][4] = {};
  Frag a_cur[2] = { load_a_frag(ap[0][0], half), load_a_frag(ap[1][0], half) };
  for (int kp = 0; kp < 8; ++kp) {
    Frag b[4];
    const unsigned short* wrow = sW1 + (cg * 64 + l16) * 256 + kp * 32 + half * 8;
#pragma unroll
    for (int j = 0; j < 4; ++j) {
      const v4u* wp = (const v4u*)(wrow + j * (16 * 256));
      b[j].q[0] = wp[0];
      b[j].q[1] = wp[2];
    }
    Frag a_next[2];
    if (kp < 7) {
      int kn = kp + 1;
#pragma unroll
      for (int mf = 0; mf < 2; ++mf) {
        const float* abase = (kn < 4 ? ap[mf][0] : ap[mf][1]) + (kn & 3) * 32;
        a_next[mf] = load_a_frag(abase, half);
      }
    }
#pragma unroll
    for (int mf = 0; mf < 2; ++mf)
#pragma unroll
      for (int j = 0; j < 4; ++j)
        acc[mf][j] = __builtin_amdgcn_wmma_f32_16x16x32_bf16(
            false, a_cur[mf].v, false, b[j].v, (short)0, acc[mf][j], false, false);
    a_cur[0] = a_next[0];
    a_cur[1] = a_next[1];
  }

#pragma unroll
  for (int mf = 0; mf < 2; ++mf)
#pragma unroll
    for (int j = 0; j < 4; ++j) {
      int col = cg * 64 + j * 16 + l16;
      float b1v = nb1[col];
#pragma unroll
      for (int v = 0; v < 8; ++v) {
        float x = acc[mf][j][v] + b1v;
        x = x > 0.f ? x : 0.f;
        int m = mg * 32 + mf * 16 + half * 8 + v;
        sH[m * 128 + col] = f2bf_fast(x);
      }
    }
  __syncthreads();

  // -------- GEMM2: [128 x 128] x [128 x 128] --------
  v8f acc2[2][4] = {};
  const unsigned short* hrow0 = sH + (mg * 32 + l16) * 128;
  const unsigned short* hrow1 = hrow0 + 16 * 128;
  for (int kp = 0; kp < 4; ++kp) {
    Frag b[4];
    const unsigned short* wrow = sW2 + (cg * 64 + l16) * 128 + kp * 32 + half * 8;
#pragma unroll
    for (int j = 0; j < 4; ++j) {
      const v4u* wp = (const v4u*)(wrow + j * (16 * 128));
      b[j].q[0] = wp[0];
      b[j].q[1] = wp[2];
    }
    Frag a[2];
#pragma unroll
    for (int v = 0; v < 8; ++v) {
      int ko = kp * 32 + ((v & 4) << 2) + half * 8 + ((v & 3) << 1);
      a[0].u[v] = *(const unsigned int*)(hrow0 + ko);
      a[1].u[v] = *(const unsigned int*)(hrow1 + ko);
    }
#pragma unroll
    for (int mf = 0; mf < 2; ++mf)
#pragma unroll
      for (int j = 0; j < 4; ++j)
        acc2[mf][j] = __builtin_amdgcn_wmma_f32_16x16x32_bf16(
            false, a[mf].v, false, b[j].v, (short)0, acc2[mf][j], false, false);
  }

  // -------- epilogue --------
  float mult = log1pf(expf(rm_w[0]));
#pragma unroll
  for (int mf = 0; mf < 2; ++mf)
#pragma unroll
    for (int j = 0; j < 4; ++j) {
      int col = cg * 64 + j * 16 + l16;
      float b2v = nb2[col];
#pragma unroll
      for (int v = 0; v < 8; ++v) {
        int n = mbase + mf * 16 + half * 8 + v;
        if (n < NN) {
          float val = acc2[mf][j][v] + b2v;
          size_t off = (size_t)n * DD + col;
          nodes_out[off] = nodes[off] + mult * val;
        }
      }
    }
}

extern "C" void kernel_launch(void* const* d_in, const int* in_sizes, int n_in,
                              void* d_out, int out_size, void* d_ws, size_t ws_size,
                              hipStream_t stream) {
  const float* nodes     = (const float*)d_in[0];
  const float* edges     = (const float*)d_in[1];
  const int*   receivers = (const int*)  d_in[2];
  const int*   senders   = (const int*)  d_in[3];
  // d_in[4] = global_latent: unused by the reference computation
  const float* eW1 = (const float*)d_in[5];
  const float* eb1 = (const float*)d_in[6];
  const float* eW2 = (const float*)d_in[7];
  const float* eb2 = (const float*)d_in[8];
  const float* nW1 = (const float*)d_in[9];
  const float* nb1 = (const float*)d_in[10];
  const float* nW2 = (const float*)d_in[11];
  const float* nb2 = (const float*)d_in[12];
  const float* rm_w = (const float*)d_in[13];

  float* nodes_out = (float*)d_out;                       // [NN, 128]
  float* edges_out = nodes_out + (size_t)NN * DD;         // [NE, 128]

  float* adj = (float*)d_ws;                              // [NN, 128] scratch
  unsigned short* wbf = (unsigned short*)((char*)d_ws + ADJ_FLOATS * sizeof(float));

  hipMemsetAsync(adj, 0, ADJ_FLOATS * sizeof(float), stream);

  prep_weights_kernel<<<120, 256, 0, stream>>>(eW1, eW2, nW1, nW2, wbf);

  const size_t sh_edge = (size_t)(128 * 384 + 128 * 128 + 128 * 128) * sizeof(unsigned short); // 160 KB
  const int eblocks = (NE + 127) / 128;
  edge_mlp_kernel<<<eblocks, 256, sh_edge, stream>>>(
      nodes, edges, receivers, senders, wbf, eb1, eb2, rm_w, edges_out, adj);

  const size_t sh_node = (size_t)(128 * 256 + 128 * 128 + 128 * 128) * sizeof(unsigned short); // 128 KB
  const int nblocks = (NN + 127) / 128;
  node_mlp_kernel<<<nblocks, 256, sh_node, stream>>>(
      nodes, adj, wbf, nb1, nb2, rm_w, nodes_out);
}
